// TRANSFORMER_88184268522189
// MI455X (gfx1250) — compile-verified, matrix operand order checked by reading, not run on hardware
//
#include <hip/hip_runtime.h>
#include <hip/hip_bf16.h>
#include <cstdint>

typedef __attribute__((ext_vector_type(16))) __bf16 v16bf;
typedef __attribute__((ext_vector_type(8)))  __bf16 v8bf;
typedef __attribute__((ext_vector_type(4)))  __bf16 v4bf;
typedef __attribute__((ext_vector_type(8)))  float  v8f;

#define DEV __device__ __forceinline__

DEV v8f vzero8() {
  v8f z;
#pragma unroll
  for (int i = 0; i < 8; ++i) z[i] = 0.0f;
  return z;
}

DEV v8f wmma_bf16(v16bf a, v16bf b, v8f c) {
  return __builtin_amdgcn_wmma_f32_16x16x32_bf16(false, a, false, b, (short)0, c, false, false);
}

// load 8 consecutive f32 via two float4s (16B-aligned)
DEV void load8f(const float* __restrict__ p, float o[8]) {
  float4 a = ((const float4*)p)[0];
  float4 b = ((const float4*)p)[1];
  o[0] = a.x; o[1] = a.y; o[2] = a.z; o[3] = a.w;
  o[4] = b.x; o[5] = b.y; o[6] = b.z; o[7] = b.w;
}

// bf16 fragment from bf16 row (16B aligned, kstart%8==0): two 16B loads, no converts
DEV v16bf frag_bf16(const __bf16* __restrict__ row, int kstart) {
  v8bf lo = *(const v8bf*)(row + kstart);
  v8bf hi = *(const v8bf*)(row + kstart + 16);
  return __builtin_shufflevector(lo, hi, 0, 1, 2, 3, 4, 5, 6, 7, 8, 9, 10, 11, 12, 13, 14, 15);
}

// ---------------------------------------------------------------------------
// bf16 GEMM: Y[MxO] = Xb[MxK] @ Wb[OxK]^T + bias. K%32==0, M%128==0.
// Tile 128x128 (8 waves: 4 row x 2 col, wave = 32x64). Columns >= O are
// clamped on load and guarded on store (needed only for O=32 embedding).
// OUT16 selects bf16 output (for XQ/XV feeding attention).
// ---------------------------------------------------------------------------
template <bool OUT16>
__global__ void __launch_bounds__(256) wmma_gemm_bf16_t(
    const __bf16* __restrict__ X, const __bf16* __restrict__ W,
    const float* __restrict__ bias, float* __restrict__ Yf, __bf16* __restrict__ Yh,
    int M, int K, int O)
{
  int lane = threadIdx.x & 31, wave = threadIdx.x >> 5;
  int rw = wave >> 1, cw = wave & 1;
  int rowbase = blockIdx.x * 128 + rw * 32;
  int colbase = blockIdx.y * 128 + cw * 64;
  int mlane = lane & 15;
  int klo = (lane & 16) ? 8 : 0;

  v8f acc[2][4];
#pragma unroll
  for (int i = 0; i < 2; ++i)
#pragma unroll
    for (int j = 0; j < 4; ++j) acc[i][j] = vzero8();

  const __bf16* xr0 = X + (size_t)(rowbase + mlane) * K;
  const __bf16* xr1 = X + (size_t)(rowbase + 16 + mlane) * K;
  const __bf16* wr[4];
#pragma unroll
  for (int j = 0; j < 4; ++j) {
    int n = colbase + j * 16 + mlane;
    if (n >= O) n = O - 1;
    wr[j] = W + (size_t)n * K;
  }

  for (int k0 = 0; k0 < K; k0 += 32) {
    v16bf a0 = frag_bf16(xr0, k0 + klo);
    v16bf a1 = frag_bf16(xr1, k0 + klo);
#pragma unroll
    for (int j = 0; j < 4; ++j) {
      v16bf b = frag_bf16(wr[j], k0 + klo);
      acc[0][j] = wmma_bf16(a0, b, acc[0][j]);
      acc[1][j] = wmma_bf16(a1, b, acc[1][j]);
    }
  }

  int roff = (lane & 16) ? 8 : 0;
#pragma unroll
  for (int i = 0; i < 2; ++i)
#pragma unroll
    for (int j = 0; j < 4; ++j) {
      int col = colbase + j * 16 + mlane;
      if (col < O) {
        float bv = bias[col];
#pragma unroll
        for (int r = 0; r < 8; ++r) {
          size_t idx = (size_t)(rowbase + i * 16 + r + roff) * O + col;
          if (OUT16) Yh[idx] = (__bf16)(acc[i][j][r] + bv);
          else       Yf[idx] = acc[i][j][r] + bv;
        }
      }
    }
}

// f32 -> bf16 conversion (n % 4 == 0)
__global__ void __launch_bounds__(256) cvt_f32_bf16(
    const float* __restrict__ src, __bf16* __restrict__ dst, long long n)
{
  long long n4 = n >> 2;
  for (long long i = (long long)blockIdx.x * blockDim.x + threadIdx.x; i < n4;
       i += (long long)gridDim.x * blockDim.x) {
    float4 v = ((const float4*)src)[i];
    v4bf o;
    o[0] = (__bf16)v.x; o[1] = (__bf16)v.y; o[2] = (__bf16)v.z; o[3] = (__bf16)v.w;
    ((v4bf*)dst)[i] = o;
  }
}

// f32 [R][K] -> bf16 [R][KP], zero-padded columns K..KP-1
__global__ void __launch_bounds__(256) pad_cvt_bf16(
    const float* __restrict__ src, __bf16* __restrict__ dst, int R, int K, int KP)
{
  long long total = (long long)R * KP;
  for (long long idx = (long long)blockIdx.x * blockDim.x + threadIdx.x; idx < total;
       idx += (long long)gridDim.x * blockDim.x) {
    int r = (int)(idx / KP), k = (int)(idx % KP);
    float v = src[(size_t)r * K + (k < K ? k : K - 1)];
    dst[idx] = (__bf16)(k < K ? v : 0.0f);
  }
}

// ---------------------------------------------------------------------------
// Edge-feature run of 8 consecutive k (k % 8 == 0, compile-time after unroll).
// ---------------------------------------------------------------------------
template <int STAGE>
DEV void feat_run8(int n, int nbr, int eidx, int k, float o[8],
                   const float* __restrict__ x0, const float* __restrict__ x1,
                   const float* __restrict__ x2, const float* __restrict__ nbrf)
{
  if (STAGE == 1) {                       // [x0[n]|x0[nbr]|nbr_fea] K=105 (pad 128)
    if (k < 32) load8f(x0 + (size_t)n * 32 + k, o);
    else if (k < 64) load8f(x0 + (size_t)nbr * 32 + (k - 32), o);
    else {
#pragma unroll
      for (int j = 0; j < 8; ++j) {
        int kk = k - 64 + j;
        float v = nbrf[(size_t)eidx * 41 + (kk < 41 ? kk : 40)];
        o[j] = (kk < 41) ? v : 0.0f;
      }
    }
  } else if (STAGE == 2) {                // [x1[n] | x1[nbr]-x1[n]] K=128
    if (k < 64) load8f(x1 + (size_t)n * 64 + k, o);
    else {
      float a[8], c[8];
      load8f(x1 + (size_t)nbr * 64 + (k - 64), a);
      load8f(x1 + (size_t)n * 64 + (k - 64), c);
#pragma unroll
      for (int j = 0; j < 8; ++j) o[j] = a[j] - c[j];
    }
  } else {                                // xc=[x1|x2]; [xc | xc_nbr - xc] K=384
    if (k < 192) {
      if (k < 64) load8f(x1 + (size_t)n * 64 + k, o);
      else load8f(x2 + (size_t)n * 128 + (k - 64), o);
    } else {
      int kk = k - 192;
      float a[8], c[8];
      if (kk < 64) {
        load8f(x1 + (size_t)nbr * 64 + kk, a);
        load8f(x1 + (size_t)n * 64 + kk, c);
      } else {
        load8f(x2 + (size_t)nbr * 128 + (kk - 64), a);
        load8f(x2 + (size_t)n * 128 + (kk - 64), c);
      }
#pragma unroll
      for (int j = 0; j < 8; ++j) o[j] = a[j] - c[j];
    }
  }
}

template <int STAGE>
DEV v16bf frag_edge(int n, int m, int nbr, int kstart,
                    const float* __restrict__ x0, const float* __restrict__ x1,
                    const float* __restrict__ x2, const float* __restrict__ nbrf)
{
  float lo[8], hi[8];
  int eidx = n * 12 + (m < 12 ? m : 11);
  feat_run8<STAGE>(n, nbr, eidx, kstart, lo, x0, x1, x2, nbrf);
  feat_run8<STAGE>(n, nbr, eidx, kstart + 16, hi, x0, x1, x2, nbrf);
  float sc = (m < 12) ? 1.0f : 0.0f;     // zero padded edge rows without branching
  v16bf v;
#pragma unroll
  for (int j = 0; j < 8; ++j) {
    v[j]     = (__bf16)(lo[j] * sc);
    v[j + 8] = (__bf16)(hi[j] * sc);
  }
  return v;
}

// ---------------------------------------------------------------------------
// Edge GEMM: one 16x16 C tile = one atom (16 padded edge rows). Emits per-atom
// channel max/min over the 12 real edges + per-channel sum/sumsq stats.
// Block = 8 waves (4 row x 2 col): 8 atoms x 64 channels. K loop fully unrolled.
// ---------------------------------------------------------------------------
template <int STAGE, int KT, int O>
__global__ void __launch_bounds__(256) edge_conv_wmma(
    const float* __restrict__ x0, const float* __restrict__ x1, const float* __restrict__ x2,
    const int* __restrict__ nbr_idx, const float* __restrict__ nbrf,
    const __bf16* __restrict__ Wb,
    float* __restrict__ hmax, float* __restrict__ hmin, float* __restrict__ stats)
{
  __shared__ float lsum[64], lssq[64];
  if (threadIdx.x < 64) { lsum[threadIdx.x] = 0.0f; lssq[threadIdx.x] = 0.0f; }
  __syncthreads();

  int lane = threadIdx.x & 31, wave = threadIdx.x >> 5;
  int rw = wave >> 1, cw = wave & 1;
  int atom0 = blockIdx.x * 8 + rw * 2;
  int colbase = blockIdx.y * 64 + cw * 32;
  int mlane = lane & 15;
  int klo = (lane & 16) ? 8 : 0;
  bool hi = (lane & 16) != 0;

  int nb0 = (mlane < 12) ? nbr_idx[atom0 * 12 + mlane] : 0;
  int nb1 = (mlane < 12) ? nbr_idx[(atom0 + 1) * 12 + mlane] : 0;

  v8f acc[2][2];
#pragma unroll
  for (int i = 0; i < 2; ++i)
#pragma unroll
    for (int j = 0; j < 2; ++j) acc[i][j] = vzero8();

  const __bf16* wb0 = Wb + (size_t)(colbase + mlane) * KT;
  const __bf16* wb1 = Wb + (size_t)(colbase + 16 + mlane) * KT;

#pragma unroll
  for (int k0 = 0; k0 < KT; k0 += 32) {
    v16bf a0 = frag_edge<STAGE>(atom0,     mlane, nb0, k0 + klo, x0, x1, x2, nbrf);
    v16bf a1 = frag_edge<STAGE>(atom0 + 1, mlane, nb1, k0 + klo, x0, x1, x2, nbrf);
    v16bf b0 = frag_bf16(wb0, k0 + klo);
    v16bf b1 = frag_bf16(wb1, k0 + klo);
    acc[0][0] = wmma_bf16(a0, b0, acc[0][0]);
    acc[0][1] = wmma_bf16(a0, b1, acc[0][1]);
    acc[1][0] = wmma_bf16(a1, b0, acc[1][0]);
    acc[1][1] = wmma_bf16(a1, b1, acc[1][1]);
  }

  // lanes<16 hold rows 0..7 (all real), lanes>=16 rows 8..15 (only 8..11 real)
  float jsum[2] = {0.0f, 0.0f}, jssq[2] = {0.0f, 0.0f};
#pragma unroll
  for (int i = 0; i < 2; ++i) {
    int n = atom0 + i;
#pragma unroll
    for (int j = 0; j < 2; ++j) {
      float mx = -3.4e38f, mn = 3.4e38f, s = 0.0f, q = 0.0f;
#pragma unroll
      for (int r = 0; r < 8; ++r) {
        float v = acc[i][j][r];
        bool valid = hi ? (r < 4) : true;
        if (valid) { mx = fmaxf(mx, v); mn = fminf(mn, v); s += v; q += v * v; }
      }
      mx = fmaxf(mx, __shfl_xor(mx, 16));
      mn = fminf(mn, __shfl_xor(mn, 16));
      s += __shfl_xor(s, 16);
      q += __shfl_xor(q, 16);
      jsum[j] += s; jssq[j] += q;
      if (!hi) {
        int col = colbase + j * 16 + mlane;
        hmax[(size_t)n * O + col] = mx;
        hmin[(size_t)n * O + col] = mn;
      }
    }
  }
  if (!hi) {
#pragma unroll
    for (int j = 0; j < 2; ++j) {
      int lcol = cw * 32 + j * 16 + mlane;
      atomicAdd(&lsum[lcol], jsum[j]);
      atomicAdd(&lssq[lcol], jssq[j]);
    }
  }
  __syncthreads();
  if (threadIdx.x < 64) {
    int gcol = blockIdx.y * 64 + threadIdx.x;
    atomicAdd(&stats[gcol],     lsum[threadIdx.x]);
    atomicAdd(&stats[O + gcol], lssq[threadIdx.x]);
  }
}

// conv finalize: relu(BN(H)), H = hmax if slope>=0 else hmin. O power of two.
__global__ void __launch_bounds__(256) conv_finalize(
    const float* __restrict__ hmax, const float* __restrict__ hmin,
    const float* __restrict__ stats, const float* __restrict__ g, const float* __restrict__ b,
    float* __restrict__ out, long long total, int O, float invCnt)
{
  for (long long idx = (long long)blockIdx.x * blockDim.x + threadIdx.x; idx < total;
       idx += (long long)gridDim.x * blockDim.x) {
    int o = (int)idx & (O - 1);
    float mean = stats[o] * invCnt;
    float var  = stats[O + o] * invCnt - mean * mean;
    float inv  = rsqrtf(var + 1e-5f);
    float gv   = g[o];
    float H    = (gv >= 0.0f) ? hmax[idx] : hmin[idx];
    out[idx]   = fmaxf(gv * (H - mean) * inv + b[o], 0.0f);
  }
}

// column stats (sum, sumsq): block (64,4), grid (ceil(O/64), Sy)
__global__ void col_stats(const float* __restrict__ Y, long long M, int O, float* __restrict__ stats)
{
  __shared__ float ls[4][64], lq[4][64];
  int c = blockIdx.x * 64 + threadIdx.x;
  float s = 0.0f, q = 0.0f;
  if (c < O) {
    for (long long r = (long long)blockIdx.y * 4 + threadIdx.y; r < M;
         r += (long long)gridDim.y * 4) {
      float v = Y[r * O + c];
      s += v; q += v * v;
    }
  }
  ls[threadIdx.y][threadIdx.x] = s;
  lq[threadIdx.y][threadIdx.x] = q;
  __syncthreads();
  if (threadIdx.y == 0 && c < O) {
    float ss = ls[0][threadIdx.x] + ls[1][threadIdx.x] + ls[2][threadIdx.x] + ls[3][threadIdx.x];
    float qq = lq[0][threadIdx.x] + lq[1][threadIdx.x] + lq[2][threadIdx.x] + lq[3][threadIdx.x];
    atomicAdd(&stats[c], ss);
    atomicAdd(&stats[O + c], qq);
  }
}

// BN apply (+relu, +skip added AFTER relu). O power of two.
__global__ void __launch_bounds__(256) bn_apply(
    const float* __restrict__ U, const float* __restrict__ stats,
    const float* __restrict__ g, const float* __restrict__ b,
    const float* __restrict__ skip, float* __restrict__ Y,
    long long total, int O, float invCnt, int doRelu)
{
  for (long long idx = (long long)blockIdx.x * blockDim.x + threadIdx.x; idx < total;
       idx += (long long)gridDim.x * blockDim.x) {
    int o = (int)idx & (O - 1);
    float mean = stats[o] * invCnt;
    float var  = stats[O + o] * invCnt - mean * mean;
    float inv  = rsqrtf(var + 1e-5f);
    float v = g[o] * (U[idx] - mean) * inv + b[o];
    if (doRelu) v = fmaxf(v, 0.0f);
    if (skip) v += skip[idx];
    Y[idx] = v;
  }
}

__global__ void zero_f32(float* __restrict__ p, long long n)
{
  for (long long i = (long long)blockIdx.x * blockDim.x + threadIdx.x; i < n;
       i += (long long)gridDim.x * blockDim.x) p[i] = 0.0f;
}

// ---------------------------------------------------------------------------
// Self-attention: one workgroup per 128-atom block.
// q (bf16 64KB) and v (bf16 128KB) staged to LDS with global_load_async_to_lds
// (ASYNCcnt path); energy = q q^T / 16 (WMMA) -> col softmax (axis=-2) ->
// row renorm (+1e-9) -> x_r = att @ v (WMMA, B from LDS) -> T = X - x_r.
// LDS = 65536 + 128*SA_EP*4 + 131072 = 264192 bytes (< 320KB/WGP).
// ---------------------------------------------------------------------------
#define SA_EP 132
#define SA_SMEM_BYTES ((size_t)65536 + (size_t)128 * SA_EP * 4 + (size_t)131072)

__global__ void __launch_bounds__(256) sa_attention(
    const float* __restrict__ X, const __bf16* __restrict__ XQb,
    const __bf16* __restrict__ XVb, float* __restrict__ T)
{
  extern __shared__ char smem[];
  __bf16* qbf = (__bf16*)smem;                                    // 128 x 256
  float*  e   = (float*)(smem + 65536);                           // 128 x SA_EP
  __bf16* vbf = (__bf16*)(smem + 65536 + (size_t)128 * SA_EP * 4); // 128 x 512

  int base = blockIdx.x * 128;

  { // async raw-copy staging: 16B per lane per issue, 48 issues < ASYNCcnt cap
    unsigned lq = (unsigned)(uintptr_t)qbf;
    uint64_t gq = (uint64_t)(uintptr_t)(XQb + (size_t)base * 256);
#pragma unroll
    for (int it = 0; it < 16; ++it) {       // 64KB / (256 thr * 16B)
      int chunk = it * 256 + (int)threadIdx.x;
      unsigned la = lq + chunk * 16;
      uint64_t ga = gq + (uint64_t)chunk * 16;
      asm volatile("global_load_async_to_lds_b128 %0, %1, off"
                   :: "v"(la), "v"(ga) : "memory");
    }
    unsigned lv = (unsigned)(uintptr_t)vbf;
    uint64_t gv = (uint64_t)(uintptr_t)(XVb + (size_t)base * 512);
#pragma unroll
    for (int it = 0; it < 32; ++it) {       // 128KB / (256 thr * 16B)
      int chunk = it * 256 + (int)threadIdx.x;
      unsigned la = lv + chunk * 16;
      uint64_t ga = gv + (uint64_t)chunk * 16;
      asm volatile("global_load_async_to_lds_b128 %0, %1, off"
                   :: "v"(la), "v"(ga) : "memory");
    }
    asm volatile("s_wait_asynccnt 0" ::: "memory");
  }
  __syncthreads();

  int lane = threadIdx.x & 31, wave = threadIdx.x >> 5;
  int mlane = lane & 15;
  int klo  = (lane & 16) ? 8 : 0;
  int roff = (lane & 16) ? 8 : 0;
  int rowb = wave * 16;

  { // ---- energy = q q^T / 16 ----
    v8f acc[8];
#pragma unroll
    for (int ct = 0; ct < 8; ++ct) acc[ct] = vzero8();
    const __bf16* arow = qbf + (size_t)(rowb + mlane) * 256;
    for (int k0 = 0; k0 < 256; k0 += 32) {
      v16bf a = frag_bf16(arow, k0 + klo);
#pragma unroll
      for (int ct = 0; ct < 8; ++ct) {
        v16bf b = frag_bf16(qbf + (size_t)(ct * 16 + mlane) * 256, k0 + klo);
        acc[ct] = wmma_bf16(a, b, acc[ct]);
      }
    }
#pragma unroll
    for (int ct = 0; ct < 8; ++ct)
#pragma unroll
      for (int r = 0; r < 8; ++r)
        e[(size_t)(rowb + r + roff) * SA_EP + ct * 16 + mlane] = acc[ct][r] * 0.0625f;
  }
  __syncthreads();

  if (threadIdx.x < 128) { // softmax over axis=-2 (down columns)
    int j = threadIdx.x;
    float mx = -1e30f;
    for (int i = 0; i < 128; ++i) mx = fmaxf(mx, e[(size_t)i * SA_EP + j]);
    float s = 0.0f;
    for (int i = 0; i < 128; ++i) {
      float ex = __expf(e[(size_t)i * SA_EP + j] - mx);
      e[(size_t)i * SA_EP + j] = ex;
      s += ex;
    }
    float inv = 1.0f / s;
    for (int i = 0; i < 128; ++i) e[(size_t)i * SA_EP + j] *= inv;
  }
  __syncthreads();

  if (threadIdx.x < 128) { // row renorm: att / (1e-9 + row sum)
    int i = threadIdx.x;
    float s = 0.0f;
    for (int j = 0; j < 128; ++j) s += e[(size_t)i * SA_EP + j];
    float inv = 1.0f / (1e-9f + s);
    for (int j = 0; j < 128; ++j) e[(size_t)i * SA_EP + j] *= inv;
  }
  __syncthreads();

  // ---- x_r = att @ v (B from LDS) ; T = X - x_r ----
  const float* arow = e + (size_t)(rowb + mlane) * SA_EP;
  for (int cc = 0; cc < 512; cc += 64) {
    v8f acc[4];
#pragma unroll
    for (int ct = 0; ct < 4; ++ct) acc[ct] = vzero8();
    for (int j0 = 0; j0 < 128; j0 += 32) {
      float lo[8], hi[8];
      load8f(arow + j0 + klo, lo);
      load8f(arow + j0 + klo + 16, hi);
      v16bf a;
#pragma unroll
      for (int j = 0; j < 8; ++j) { a[j] = (__bf16)lo[j]; a[j + 8] = (__bf16)hi[j]; }
#pragma unroll
      for (int ct = 0; ct < 4; ++ct) {
        int col = cc + ct * 16 + mlane;
        v16bf b;
#pragma unroll
        for (int j = 0; j < 8; ++j) {
          b[j]     = vbf[(size_t)(j0 + klo + j) * 512 + col];
          b[j + 8] = vbf[(size_t)(j0 + 16 + klo + j) * 512 + col];
        }
        acc[ct] = wmma_bf16(a, b, acc[ct]);
      }
    }
#pragma unroll
    for (int ct = 0; ct < 4; ++ct) {
      int col = cc + ct * 16 + mlane;
#pragma unroll
      for (int r = 0; r < 8; ++r) {
        int row = base + rowb + r + roff;
        T[(size_t)row * 512 + col] = X[(size_t)row * 512 + col] - acc[ct][r];
      }
    }
  }
}

// mean pool over 128 atoms per block; columns = concat(y1,y2,y3)
__global__ void __launch_bounds__(256) pool_mean(
    const float* __restrict__ y1, const float* __restrict__ y2, const float* __restrict__ y3,
    float* __restrict__ pooled, int B, int n_per)
{
  int total = B * 1536;
  for (int idx = blockIdx.x * blockDim.x + threadIdx.x; idx < total;
       idx += gridDim.x * blockDim.x) {
    int b = idx / 1536, c = idx % 1536;
    const float* src; int col;
    if (c < 512)       { src = y1; col = c; }
    else if (c < 1024) { src = y2; col = c - 512; }
    else               { src = y3; col = c - 1024; }
    float s = 0.0f;
    for (int i = 0; i < n_per; ++i) s += src[(size_t)(b * n_per + i) * 512 + col];
    pooled[idx] = s * (1.0f / n_per);
  }
}

__global__ void __launch_bounds__(256) final_linear(
    const float* __restrict__ H, const float* __restrict__ w,
    const float* __restrict__ bias, float* __restrict__ out, int K)
{
  __shared__ float red[256];
  int b = blockIdx.x;
  float s = 0.0f;
  for (int k = threadIdx.x; k < K; k += blockDim.x) s += H[(size_t)b * K + k] * w[k];
  red[threadIdx.x] = s;
  __syncthreads();
  for (int st = 128; st > 0; st >>= 1) {
    if (threadIdx.x < st) red[threadIdx.x] += red[threadIdx.x + st];
    __syncthreads();
  }
  if (threadIdx.x == 0) out[b] = red[0] + bias[0];
}

// ---------------------------------------------------------------------------
extern "C" void kernel_launch(void* const* d_in, const int* in_sizes, int n_in,
                              void* d_out, int out_size, void* d_ws, size_t ws_size,
                              hipStream_t stream)
{
  (void)in_sizes; (void)n_in; (void)out_size; (void)ws_size;
  const int N = 65536, NPER = 128, B = N / NPER;

  const float* atom_fea = (const float*)d_in[0];
  const float* nbr_fea  = (const float*)d_in[1];
  const int*   nbr_idx  = (const int*)d_in[2];
#define PF(i) ((const float*)d_in[(i)])
  // flatten order: emb(4..7) c1(8..11) c2(12..15) c3(16..19)
  // sa0(20..27) sa1(28..35) sa2(36..43) l1(44..47) l2(48..51) l3(52..53)

  float* x0   = (float*)d_ws;                  // N*32
  float* x1   = x0   + (size_t)N * 32;         // N*64
  float* x2   = x1   + (size_t)N * 64;         // N*128
  float* big0 = x2   + (size_t)N * 128;        // N*512 : x3 -> y2
  float* big1 = big0 + (size_t)N * 512;        // N*512 : hmax -> y1
  float* big2 = big1 + (size_t)N * 512;        // N*512 : hmin -> y3
  float* xq   = big2 + (size_t)N * 512;        // N*256 f32 region (holds XQ bf16)
  float* xv   = xq   + (size_t)N * 256;        // N*512 (XV bf16 / pre-BN f32 scratch)
  float* t1   = xv   + (size_t)N * 512;        // N*512 : x_trans
  float* stats  = t1 + (size_t)N * 512;        // 4096
  float* pooled = stats + 4096;                // 512*1536
  float* u5 = pooled + (size_t)512 * 1536;     // 512*1024
  float* h5 = u5 + (size_t)512 * 1024;         // 512*1024
  float* u6 = h5 + (size_t)512 * 1024;         // 512*512
  float* h6 = u6 + (size_t)512 * 512;          // 512*512
  __bf16* xb = (__bf16*)(h6 + (size_t)512 * 512);       // N*512 bf16 act copies
  __bf16* wb = (__bf16*)((float*)xb + (size_t)N * 256); // up to 1024*1536 bf16
  __bf16* xqb = (__bf16*)xq;
  __bf16* xvb = (__bf16*)xv;

  (void)hipFuncSetAttribute((const void*)sa_attention,
                            hipFuncAttributeMaxDynamicSharedMemorySize,
                            (int)SA_SMEM_BYTES);

  // ---- embedding: pad K 92->96, bf16 GEMM, BN+relu -> x0 (N x 32)
  pad_cvt_bf16<<<4096, 256, 0, stream>>>(atom_fea, xb, N, 92, 96);
  pad_cvt_bf16<<<8, 256, 0, stream>>>(PF(4), wb, 32, 92, 96);
  wmma_gemm_bf16_t<false><<<dim3(N / 128, 1), 256, 0, stream>>>(
      xb, wb, PF(5), xv, nullptr, N, 96, 32);
  zero_f32<<<4, 256, 0, stream>>>(stats, 64);
  col_stats<<<dim3(1, 64), dim3(64, 4), 0, stream>>>(xv, N, 32, stats);
  bn_apply<<<4096, 256, 0, stream>>>(xv, stats, PF(6), PF(7), nullptr, x0,
                                     (long long)N * 32, 32, 1.0f / N, 1);

  // ---- conv1 (K=105 pad 128, O=64) -> x1
  pad_cvt_bf16<<<16, 256, 0, stream>>>(PF(8), wb, 64, 105, 128);
  zero_f32<<<4, 256, 0, stream>>>(stats, 128);
  edge_conv_wmma<1, 128, 64><<<dim3(N / 8, 1), 256, 0, stream>>>(
      x0, x1, x2, nbr_idx, nbr_fea, wb, big1, big2, stats);
  conv_finalize<<<4096, 256, 0, stream>>>(big1, big2, stats, PF(10), PF(11), x1,
                                          (long long)N * 64, 64, 1.0f / ((float)N * 12.0f));

  // ---- conv2 (K=128, O=128) -> x2
  cvt_f32_bf16<<<64, 256, 0, stream>>>(PF(12), wb, 128 * 128);
  zero_f32<<<4, 256, 0, stream>>>(stats, 256);
  edge_conv_wmma<2, 128, 128><<<dim3(N / 8, 2), 256, 0, stream>>>(
      x0, x1, x2, nbr_idx, nbr_fea, wb, big1, big2, stats);
  conv_finalize<<<4096, 256, 0, stream>>>(big1, big2, stats, PF(14), PF(15), x2,
                                          (long long)N * 128, 128, 1.0f / ((float)N * 12.0f));

  // ---- conv3 (K=384, O=512) -> x3 = big0
  cvt_f32_bf16<<<192, 256, 0, stream>>>(PF(16), wb, 512 * 384);
  zero_f32<<<4, 256, 0, stream>>>(stats, 1024);
  edge_conv_wmma<3, 384, 512><<<dim3(N / 8, 8), 256, 0, stream>>>(
      x0, x1, x2, nbr_idx, nbr_fea, wb, big1, big2, stats);
  conv_finalize<<<8192, 256, 0, stream>>>(big1, big2, stats, PF(18), PF(19), big0,
                                          (long long)N * 512, 512, 1.0f / ((float)N * 12.0f));

  // ---- self-attention layers
  auto run_sa = [&](const float* xin, int pb, float* yout) {
    cvt_f32_bf16<<<8192, 256, 0, stream>>>(xin, xb, (long long)N * 512);
    cvt_f32_bf16<<<128, 256, 0, stream>>>(PF(pb + 0), wb, 256 * 512);
    wmma_gemm_bf16_t<true><<<dim3(N / 128, 2), 256, 0, stream>>>(
        xb, wb, PF(pb + 1), nullptr, xqb, N, 512, 256);
    cvt_f32_bf16<<<256, 256, 0, stream>>>(PF(pb + 2), wb, 512 * 512);
    wmma_gemm_bf16_t<true><<<dim3(N / 128, 4), 256, 0, stream>>>(
        xb, wb, PF(pb + 3), nullptr, xvb, N, 512, 512);
    sa_attention<<<dim3(B), 256, SA_SMEM_BYTES, stream>>>(xin, xqb, xvb, t1);
    cvt_f32_bf16<<<8192, 256, 0, stream>>>(t1, xb, (long long)N * 512);
    cvt_f32_bf16<<<256, 256, 0, stream>>>(PF(pb + 4), wb, 512 * 512);
    wmma_gemm_bf16_t<false><<<dim3(N / 128, 4), 256, 0, stream>>>(
        xb, wb, PF(pb + 5), xv, nullptr, N, 512, 512);
    zero_f32<<<4, 256, 0, stream>>>(stats, 1024);
    col_stats<<<dim3(8, 64), dim3(64, 4), 0, stream>>>(xv, N, 512, stats);
    bn_apply<<<8192, 256, 0, stream>>>(xv, stats, PF(pb + 6), PF(pb + 7), xin, yout,
                                       (long long)N * 512, 512, 1.0f / N, 1);
  };
  run_sa(big0, 20, big1);   // y1
  run_sa(big1, 28, big0);   // y2 (x3 dead)
  run_sa(big0, 36, big2);   // y3 (hmin dead)

  // ---- pooling
  pool_mean<<<3072, 256, 0, stream>>>(big1, big0, big2, pooled, B, NPER);

  // ---- MLP head
  cvt_f32_bf16<<<768, 256, 0, stream>>>(pooled, xb, (long long)512 * 1536);
  cvt_f32_bf16<<<1536, 256, 0, stream>>>(PF(44), wb, (long long)1024 * 1536);
  wmma_gemm_bf16_t<false><<<dim3(4, 8), 256, 0, stream>>>(
      xb, wb, PF(45), u5, nullptr, 512, 1536, 1024);
  zero_f32<<<4, 256, 0, stream>>>(stats, 2048);
  col_stats<<<dim3(16, 16), dim3(64, 4), 0, stream>>>(u5, 512, 1024, stats);
  bn_apply<<<2048, 256, 0, stream>>>(u5, stats, PF(46), PF(47), nullptr, h5,
                                     (long long)512 * 1024, 1024, 1.0f / 512.0f, 1);

  cvt_f32_bf16<<<512, 256, 0, stream>>>(h5, xb, (long long)512 * 1024);
  cvt_f32_bf16<<<512, 256, 0, stream>>>(PF(48), wb, (long long)512 * 1024);
  wmma_gemm_bf16_t<false><<<dim3(4, 4), 256, 0, stream>>>(
      xb, wb, PF(49), u6, nullptr, 512, 1024, 512);
  zero_f32<<<4, 256, 0, stream>>>(stats, 1024);
  col_stats<<<dim3(8, 16), dim3(64, 4), 0, stream>>>(u6, 512, 512, stats);
  bn_apply<<<1024, 256, 0, stream>>>(u6, stats, PF(50), PF(51), nullptr, h6,
                                     (long long)512 * 512, 512, 1.0f / 512.0f, 1);

  final_linear<<<512, 256, 0, stream>>>(h6, PF(52), PF(53), (float*)d_out, 512);
#undef PF
}
